// BertClassifier_62938450755707
// MI455X (gfx1250) — matrix operand order repaired and checked by reading, not compiled, not run on hardware
//
#include <hip/hip_runtime.h>
#include <hip/hip_bf16.h>

// ---------------------------------------------------------------------------
// BertClassifier span scorer, fused for MI455X (gfx1250, wave32, WMMA bf16,
// TDM tensor_load_to_lds staging)
// ---------------------------------------------------------------------------
// h[b,i,j,:] = relu(P[b,i] + Q[b,j] + ind*w1_last + b1);  out = h @ W2 + b2
//   P = vecs @ W1[0:768],  Q = vecs @ W1[768:1536]
// then mask by span_avail and log_softmax over the (T*T) axis.
// ---------------------------------------------------------------------------

typedef __attribute__((ext_vector_type(16))) __bf16 v16bf;
typedef __attribute__((ext_vector_type(8)))  float  v8f;

#define B_   4
#define T_   128
#define D_   768
#define H_   770
#define L_   40
#define HP   800           // H padded to 25*32
#define KC1  24            // 768/32 K-chunks for projection GEMM
#define KC2  25            // 800/32 K-chunks for classifier GEMM
#define NT1  50            // 800/16 N-tiles for projection GEMM
#define MT1  32            // 512/16 M-tiles (B*T rows)
#define NT2  3             // 48/16 N-tiles for classifier (L=40 padded to 48)
#define BT   (B_ * T_)     // 512
#define TT   (T_ * T_)     // 16384
#define IW   8             // i-rows (waves) per fused workgroup

// ---- workspace layout (bytes, all 128-aligned) ----------------------------
static constexpr size_t OFF_VECS  = 0;                                   // 512*768 bf16
static constexpr size_t SZ_VECS   = (size_t)BT * D_ * 2;                 // 786432
static constexpr size_t OFF_W1P   = OFF_VECS + SZ_VECS;                  // packed frags
static constexpr size_t SZ_W1P    = (size_t)2 * NT1 * KC1 * 512 * 2;     // 2457600
static constexpr size_t OFF_W2P   = OFF_W1P + SZ_W1P;
static constexpr size_t SZ_W2P    = (size_t)NT2 * KC2 * 512 * 2;         // 76800
static constexpr size_t OFF_P     = OFF_W2P + SZ_W2P;
static constexpr size_t SZ_P      = (size_t)BT * HP * 4;                 // 1638400
static constexpr size_t OFF_Q     = OFF_P + SZ_P;
static constexpr size_t SZ_Q      = SZ_P;
static constexpr size_t OFF_W1L   = OFF_Q + SZ_Q;                        // padded w1 last row
static constexpr size_t OFF_B1P   = OFF_W1L + HP * 4;                    // padded b1
static constexpr size_t OFF_LSE   = OFF_B1P + HP * 4;                    // 160 f32

// ---------------------------------------------------------------------------
// Tensor Data Mover: 2D tile (rows x cols of f32, row stride in elements)
// from global memory to LDS.  D# per CDNA5 ISA §8 (CDNA5 ISA).
// This toolchain exposes the 6-arg builtin:
//   (uint32x4 g0, int32x8 g1, int32x4, int32x4, int32x8, i32 cpol)
// Groups 2/3 (and the trailing descriptor words) are zero for a 2D tensor.
// ---------------------------------------------------------------------------
#if __has_builtin(__builtin_amdgcn_tensor_load_to_lds) && \
    __has_builtin(__builtin_amdgcn_s_wait_tensorcnt)
#define HAVE_TDM 1
typedef unsigned int v4u __attribute__((ext_vector_type(4)));
typedef int          v8i __attribute__((ext_vector_type(8)));
typedef int          v4i __attribute__((ext_vector_type(4)));

__device__ __forceinline__ void tdm_load_2d_f32(const float* gaddr, unsigned lds_off,
                                                unsigned rows, unsigned cols,
                                                unsigned stride_elems) {
    unsigned long long ga = (unsigned long long)(uintptr_t)gaddr;
    v4u g0;
    g0[0] = 1u;                                            // count=1, user descriptor
    g0[1] = lds_off;                                       // lds_addr (bytes)
    g0[2] = (unsigned)(ga & 0xffffffffu);                  // global_addr[31:0]
    g0[3] = (unsigned)((ga >> 32) & 0x01ffffffu)           // global_addr[56:32]
            | (2u << 30);                                  // type = 2 ("image")
    v8i g1;
    g1[0] = (int)(2u << 16);                               // data_size=2 -> 4 bytes
    g1[1] = (int)((cols & 0xffffu) << 16);                 // tensor_dim0[15:0]
    g1[2] = (int)(((cols >> 16) & 0xffffu)                 // tensor_dim0[31:16]
            | ((rows & 0xffffu) << 16));                   // tensor_dim1[15:0]
    g1[3] = (int)(((rows >> 16) & 0xffffu)                 // tensor_dim1[31:16]
            | ((cols & 0xffffu) << 16));                   // tile_dim0
    g1[4] = (int)((rows & 0xffffu));                       // tile_dim1 (tile_dim2=0)
    g1[5] = (int)stride_elems;                             // tensor_dim0_stride[31:0]
    g1[6] = 0;                                             // stride[47:32], dim1_stride lo
    g1[7] = 0;                                             // dim1_stride hi
    v4i gz4  = {0, 0, 0, 0};                               // group 2 (unused, 2D)
    v4i gz4b = {0, 0, 0, 0};                               // group 3 (unused, 2D)
    v8i gz8  = {0, 0, 0, 0, 0, 0, 0, 0};                   // trailing words (unused)
    __builtin_amdgcn_tensor_load_to_lds(g0, g1, gz4, gz4b, gz8, 0);
}
#else
#define HAVE_TDM 0
#endif

// ---------------------------------------------------------------------------
// Prep: vecs = hidden[:,1:T+1,:] converted to bf16 (row-major 512 x 768)
// ---------------------------------------------------------------------------
__global__ void k_prep_vecs(const float* __restrict__ hidden, __bf16* __restrict__ vecs) {
    int idx = blockIdx.x * blockDim.x + threadIdx.x;
    if (idx >= BT * D_) return;
    int bt = idx / D_, d = idx % D_;
    int b = bt / T_, t = bt % T_;
    vecs[idx] = (__bf16)hidden[((size_t)b * (T_ + 1) + (t + 1)) * D_ + d];
}

// ---------------------------------------------------------------------------
// Prep: pack W1[0:768] and W1[768:1536] into WMMA B-fragment order.
// B-fragment layout (16x16x32 bf16): lane<16 -> col=lane,   K = kc*32 + {0..7,16..23}
//                                    lane>=16 -> col=lane-16, K = kc*32 + {8..15,24..31}
// flat: (((sel*NT1 + nt)*KC1 + kc)*32 + lane)*16 + t
// ---------------------------------------------------------------------------
__global__ void k_prep_w1(const float* __restrict__ W1, __bf16* __restrict__ w1p) {
    int idx = blockIdx.x * blockDim.x + threadIdx.x;
    if (idx >= 2 * NT1 * KC1 * 512) return;
    int t    = idx & 15;
    int lane = (idx >> 4) & 31;
    int rest = idx >> 9;
    int kc   = rest % KC1;
    int nt   = (rest / KC1) % NT1;
    int sel  = rest / (KC1 * NT1);
    int col  = nt * 16 + (lane & 15);
    int k    = kc * 32 + ((lane < 16) ? 0 : 8) + ((t < 8) ? t : t + 8);   // k < 768 always
    float v  = (col < H_) ? W1[(size_t)(sel * D_ + k) * H_ + col] : 0.0f;
    w1p[idx] = (__bf16)v;
}

// same packing for W2 (770x40 padded to 800x48)
__global__ void k_prep_w2(const float* __restrict__ W2, __bf16* __restrict__ w2p) {
    int idx = blockIdx.x * blockDim.x + threadIdx.x;
    if (idx >= NT2 * KC2 * 512) return;
    int t    = idx & 15;
    int lane = (idx >> 4) & 31;
    int rest = idx >> 9;
    int kc   = rest % KC2;
    int nt   = rest / KC2;
    int col  = nt * 16 + (lane & 15);
    int k    = kc * 32 + ((lane < 16) ? 0 : 8) + ((t < 8) ? t : t + 8);
    float v  = (col < L_ && k < H_) ? W2[(size_t)k * L_ + col] : 0.0f;
    w2p[idx] = (__bf16)v;
}

// pad w1_last (row 2*D of W1) and b1 to HP with zeros
__global__ void k_prep_small(const float* __restrict__ W1, const float* __restrict__ b1,
                             float* __restrict__ w1l, float* __restrict__ b1p) {
    int idx = blockIdx.x * blockDim.x + threadIdx.x;
    if (idx >= 2 * HP) return;
    if (idx < HP) w1l[idx]      = (idx < H_) ? W1[(size_t)(2 * D_) * H_ + idx] : 0.0f;
    else          b1p[idx - HP] = (idx - HP < H_) ? b1[idx - HP] : 0.0f;
}

// ---------------------------------------------------------------------------
// Projection GEMM: P/Q (512 x 800, f32) = vecs_bf16 (512x768) @ W1packed
// One wave per 16x16 tile. A-fragment per ISA: lane<16 -> row=lane, K-halves
// {0..7,16..23}; lane>=16 -> row=lane-16, K-halves {8..15,24..31}.
// ---------------------------------------------------------------------------
__global__ void k_proj(const __bf16* __restrict__ vecs, const __bf16* __restrict__ w1p,
                       float* __restrict__ P, float* __restrict__ Q) {
    int mt   = blockIdx.x;          // 0..31
    int nt   = blockIdx.y;          // 0..49
    int sel  = blockIdx.z;          // 0: P, 1: Q
    int lane = threadIdx.x;         // 0..31
    int koffA = (lane < 16) ? 0 : 8;

    const __bf16* Abase = vecs + (size_t)(mt * 16 + (lane & 15)) * D_;
    const __bf16* Bbase = w1p + ((size_t)(sel * NT1 + nt) * KC1) * 512;

    v8f acc = {};
    for (int kc = 0; kc < KC1; ++kc) {
        v16bf a, bf;
        const __bf16* ap = Abase + kc * 32 + koffA;
        const __bf16* bp = Bbase + (size_t)kc * 512 + lane * 16;
#pragma unroll
        for (int t = 0; t < 8; ++t) { a[t] = ap[t]; a[8 + t] = ap[16 + t]; }
#pragma unroll
        for (int t = 0; t < 16; ++t) bf[t] = bp[t];
        acc = __builtin_amdgcn_wmma_f32_16x16x32_bf16(false, a, false, bf,
                                                      (short)0, acc, false, false);
    }

    float* outp = sel ? Q : P;
    int col  = nt * 16 + (lane & 15);
    int rowb = mt * 16 + ((lane < 16) ? 0 : 8);
#pragma unroll
    for (int r = 0; r < 8; ++r) outp[(size_t)(rowb + r) * HP + col] = acc[r];
}

// ---------------------------------------------------------------------------
// Fused classifier: 8 waves per workgroup, one Q tile shared by all waves;
// wave w handles row i = itile*8 + w.  Q/P tiles staged into LDS by the
// Tensor Data Mover (wave 0), synchronized via s_wait_tensorcnt + barrier.
// Each wave builds the bf16 A-fragment of relu(h) per 32-wide K chunk and
// runs 3 WMMAs (L padded to 48).  Masked store with +b2.
// ---------------------------------------------------------------------------
__global__ void __launch_bounds__(32 * IW)
k_fused(const float* __restrict__ P, const float* __restrict__ Q,
        const float* __restrict__ w1l, const float* __restrict__ b1p,
        const __bf16* __restrict__ w2p, const float* __restrict__ b2,
        const int* __restrict__ spans, const int* __restrict__ avail,
        float* __restrict__ out) {
    __shared__ float sQ[16][HP];        // 51.2 KB
    __shared__ float sP[IW][HP];        // 25.6 KB
    __shared__ float sW[HP];
    __shared__ float sB[HP];

    int bit   = blockIdx.x;             // 0..63  (b * 16 + itile)
    int b     = bit >> 4;
    int itile = bit & 15;
    int jtile = blockIdx.y;             // 0..7
    int tid   = threadIdx.x;            // 0..255
    int wave  = tid >> 5;
    int lane  = tid & 31;
    int i     = itile * IW + wave;

    const float* Qtile = Q + (size_t)(b * T_ + jtile * 16) * HP;
    const float* Ptile = P + (size_t)(b * T_ + itile * IW) * HP;

#if HAVE_TDM
    if (tid < 32) {                     // wave 0 drives the TDM
        tdm_load_2d_f32(Qtile, (unsigned)(uintptr_t)&sQ[0][0], 16, HP, HP);
        tdm_load_2d_f32(Ptile, (unsigned)(uintptr_t)&sP[0][0], IW, HP, HP);
        __builtin_amdgcn_s_wait_tensorcnt(0);
    }
    for (int k = tid; k < HP; k += 32 * IW) { sW[k] = w1l[k]; sB[k] = b1p[k]; }
#else
    for (int x = tid; x < 16 * HP; x += 32 * IW) sQ[x / HP][x % HP] = Qtile[x];
    for (int x = tid; x < IW * HP; x += 32 * IW) sP[x / HP][x % HP] = Ptile[x];
    for (int k = tid; k < HP; k += 32 * IW) { sW[k] = w1l[k]; sB[k] = b1p[k]; }
#endif
    __syncthreads();

    // indicator for this lane's A-fragment row (j = jtile*16 + (lane&15))
    int jloc = lane & 15;
    int j_a  = jtile * 16 + jloc;
    int s0   = spans[2 * b], e0 = spans[2 * b + 1];
    bool full   = (i == s0) && (j_a == e0);
    bool inside = (s0 <= i) && (i <= j_a) && (j_a <= e0) && !full;
    float ind   = full ? 2.0f : (inside ? 1.0f : 0.0f);

    int koffA = (lane < 16) ? 0 : 8;
    v8f acc0 = {}, acc1 = {}, acc2 = {};

    for (int kc = 0; kc < KC2; ++kc) {
        int kb = kc * 32 + koffA;
        v16bf a;
#pragma unroll
        for (int t = 0; t < 8; ++t) {
            int k0 = kb + t;
            int k1 = kb + 16 + t;
            float h0 = sP[wave][k0] + sQ[jloc][k0] + ind * sW[k0] + sB[k0];
            float h1 = sP[wave][k1] + sQ[jloc][k1] + ind * sW[k1] + sB[k1];
            a[t]     = (__bf16)fmaxf(h0, 0.0f);
            a[8 + t] = (__bf16)fmaxf(h1, 0.0f);
        }
        const __bf16* bp = w2p + (size_t)kc * 512 + lane * 16;
        v16bf f0, f1, f2;
#pragma unroll
        for (int t = 0; t < 16; ++t) {
            f0[t] = bp[t];
            f1[t] = bp[(size_t)KC2 * 512 + t];
            f2[t] = bp[(size_t)2 * KC2 * 512 + t];
        }
        acc0 = __builtin_amdgcn_wmma_f32_16x16x32_bf16(false, a, false, f0,
                                                       (short)0, acc0, false, false);
        acc1 = __builtin_amdgcn_wmma_f32_16x16x32_bf16(false, a, false, f1,
                                                       (short)0, acc1, false, false);
        acc2 = __builtin_amdgcn_wmma_f32_16x16x32_bf16(false, a, false, f2,
                                                       (short)0, acc2, false, false);
    }

    // store: C layout -> row M = r + (lane<16?0:8) is j_local; col N = lane&15
    int lcol = lane & 15;
#pragma unroll
    for (int r = 0; r < 8; ++r) {
        int jr = jtile * 16 + r + ((lane < 16) ? 0 : 8);
        bool m = avail[i * T_ + jr] >= 1;
        size_t rowoff = ((size_t)b * TT + (size_t)i * T_ + jr) * L_;
        float vals[3] = {acc0[r], acc1[r], acc2[r]};
#pragma unroll
        for (int nt = 0; nt < NT2; ++nt) {
            int l = nt * 16 + lcol;
            if (l < L_) out[rowoff + l] = m ? (vals[nt] + b2[l]) : 0.0f;
        }
    }
}

// ---------------------------------------------------------------------------
// logsumexp over axis 1 (T*T = 16384) per (b, l): two-pass max + sum(exp)
// ---------------------------------------------------------------------------
__global__ void k_lse(const float* __restrict__ out, float* __restrict__ lse) {
    __shared__ float red[256];
    int b = blockIdx.x / L_, l = blockIdx.x % L_;
    int tid = threadIdx.x;
    const float* base = out + (size_t)b * TT * L_ + l;

    float m = -__builtin_inff();
    for (int p = tid; p < TT; p += 256) m = fmaxf(m, base[(size_t)p * L_]);
    red[tid] = m;
    __syncthreads();
    for (int w = 128; w > 0; w >>= 1) {
        if (tid < w) red[tid] = fmaxf(red[tid], red[tid + w]);
        __syncthreads();
    }
    float M = red[0];
    __syncthreads();

    float s = 0.0f;
    for (int p = tid; p < TT; p += 256) s += __expf(base[(size_t)p * L_] - M);
    red[tid] = s;
    __syncthreads();
    for (int w = 128; w > 0; w >>= 1) {
        if (tid < w) red[tid] += red[tid + w];
        __syncthreads();
    }
    if (tid == 0) lse[blockIdx.x] = M + __logf(red[0]);
}

__global__ void k_apply(float* __restrict__ out, const float* __restrict__ lse) {
    int idx = blockIdx.x * blockDim.x + threadIdx.x;
    if (idx >= B_ * TT * L_) return;
    int b = idx / (TT * L_);
    int l = idx % L_;
    out[idx] -= lse[b * L_ + l];
}

// ---------------------------------------------------------------------------
extern "C" void kernel_launch(void* const* d_in, const int* in_sizes, int n_in,
                              void* d_out, int out_size, void* d_ws, size_t ws_size,
                              hipStream_t stream) {
    const float* hidden = (const float*)d_in[0];
    const float* W1     = (const float*)d_in[1];
    const float* b1     = (const float*)d_in[2];
    const float* W2     = (const float*)d_in[3];
    const float* b2     = (const float*)d_in[4];
    const int*   spans  = (const int*)d_in[5];
    const int*   avail  = (const int*)d_in[6];
    float*       out    = (float*)d_out;

    char* ws = (char*)d_ws;
    __bf16* vecs = (__bf16*)(ws + OFF_VECS);
    __bf16* w1p  = (__bf16*)(ws + OFF_W1P);
    __bf16* w2p  = (__bf16*)(ws + OFF_W2P);
    float*  P    = (float*)(ws + OFF_P);
    float*  Q    = (float*)(ws + OFF_Q);
    float*  w1l  = (float*)(ws + OFF_W1L);
    float*  b1p  = (float*)(ws + OFF_B1P);
    float*  lse  = (float*)(ws + OFF_LSE);

    k_prep_vecs<<<(BT * D_ + 255) / 256, 256, 0, stream>>>(hidden, vecs);
    k_prep_w1<<<(2 * NT1 * KC1 * 512 + 255) / 256, 256, 0, stream>>>(W1, w1p);
    k_prep_w2<<<(NT2 * KC2 * 512 + 255) / 256, 256, 0, stream>>>(W2, w2p);
    k_prep_small<<<(2 * HP + 255) / 256, 256, 0, stream>>>(W1, b1, w1l, b1p);

    k_proj<<<dim3(MT1, NT1, 2), 32, 0, stream>>>(vecs, w1p, P, Q);

    k_fused<<<dim3(BT / IW, T_ / 16), 32 * IW, 0, stream>>>(P, Q, w1l, b1p, w2p, b2,
                                                            spans, avail, out);

    k_lse<<<B_ * L_, 256, 0, stream>>>(out, lse);
    k_apply<<<(B_ * TT * L_ + 255) / 256, 256, 0, stream>>>(out, lse);
}
